// VTop_73899207295592
// MI455X (gfx1250) — matrix operand order repaired and checked by reading, not compiled, not run on hardware
//
#include <hip/hip_runtime.h>
#include <stdint.h>

typedef _Float16 h16;
typedef __attribute__((ext_vector_type(2)))  _Float16 v2h;
typedef __attribute__((ext_vector_type(4)))  _Float16 v4h;
typedef __attribute__((ext_vector_type(8)))  _Float16 v8h;
typedef __attribute__((ext_vector_type(16))) _Float16 v16h;
typedef __attribute__((ext_vector_type(4)))  float    v4f;
typedef __attribute__((ext_vector_type(8)))  float    v8f;
typedef __attribute__((ext_vector_type(4)))  uint32_t v4u;
typedef __attribute__((ext_vector_type(4)))  int32_t  v4i;
typedef __attribute__((ext_vector_type(8)))  int32_t  v8i;

static constexpr int B_ = 4;
static constexpr int H_ = 12;
static constexpr int P_ = 1568;
static constexpr int E_ = 64;     // head dim
static constexpr int T_ = 8;      // FRAME_T
static constexpr int D_ = 196;    // P / FRAME_T
static constexpr int ROWS = 32;   // rows (p values) per single-wave workgroup
static constexpr int PBLK = P_ / ROWS;   // 49
static constexpr int ASTR = 232;  // halves per A-buffer row (224 padded K + 8 slack, 16B aligned)
static constexpr float INV_COUNT = 1.0f / 38535168.0f;  // 1 / (B*H*P*T*E)

#if __has_builtin(__builtin_amdgcn_tensor_load_to_lds) && __has_builtin(__builtin_amdgcn_s_wait_tensorcnt)
#define USE_TDM 1
#else
#define USE_TDM 0
#endif

#if USE_TDM
// Build a 2D TDM descriptor: tile = 196 x 32 f32, row pitch 1568 floats, dest LDS.
__device__ __forceinline__ void tdm_load_tile(const void* gptr, uint32_t lds_byte_off) {
  v4u g0;
  uint64_t ga = (uint64_t)(uintptr_t)gptr;
  g0[0] = 1u;                                   // count=1 (valid), load, not restore, gather off
  g0[1] = lds_byte_off;                         // lds_addr
  g0[2] = (uint32_t)ga;                         // global_addr[31:0]
  g0[3] = (uint32_t)((ga >> 32) & 0x01FFFFFFull) | (2u << 30);  // addr[56:32] | type=2
  v8i g1;
  g1[0] = (int)(2u << 16);                      // data_size=4B; no multicast/pad/iterate
  g1[1] = (int)(((uint32_t)D_) << 16);          // tensor_dim0[15:0] = 196
  g1[2] = (int)(((uint32_t)ROWS) << 16);        // tensor_dim0 hi=0 | tensor_dim1[15:0]=32
  g1[3] = (int)(((uint32_t)D_) << 16);          // tensor_dim1 hi=0 | tile_dim0=196
  g1[4] = (int)ROWS;                            // tile_dim1=32, tile_dim2=0
  g1[5] = (int)P_;                              // tensor_dim0_stride = 1568 elements
  g1[6] = 0;
  g1[7] = 0;
  v4i gz = {0, 0, 0, 0};
#if defined(__clang_major__) && (__clang_major__ >= 23)
  v8i gz8 = {0, 0, 0, 0, 0, 0, 0, 0};
  __builtin_amdgcn_tensor_load_to_lds(g0, g1, gz, gz, gz8, 0);
#else
  __builtin_amdgcn_tensor_load_to_lds(g0, g1, gz, gz, 0);
#endif
}
#endif

// Pre-pass: vh[bh][t][e][d] = (f16) v[bh][d*8 + t][e]  (puts K-pairs contiguous for B tiles)
__global__ __launch_bounds__(256) void vconvert(const float* __restrict__ v,
                                                h16* __restrict__ vh) {
  int idx = blockIdx.x * 256 + threadIdx.x;
  const int total = (B_ * H_) * T_ * E_ * D_;
  if (idx >= total) return;
  int d = idx % D_;
  int r = idx / D_;
  int e = r % E_; r /= E_;
  int t = r % T_; r /= T_;
  int bh = r;
  vh[idx] = (h16)v[((size_t)bh * P_ + (size_t)d * T_ + t) * E_ + e];
}

__global__ __launch_bounds__(32) void vtop_fused(
    const float* __restrict__ att_s, const float* __restrict__ att_t,
    const h16* __restrict__ vh_s, const h16* __restrict__ vh_t,
    float* __restrict__ out) {
  __shared__ __align__(16) float logits[ROWS * D_];   // 25088 B, TDM destination
  __shared__ __align__(16) h16   abuf[ROWS * ASTR];   // 14848 B, WMMA A operand

  const int lane = threadIdx.x;
  const int pblk = blockIdx.x;   // 0..48
  const int tg   = blockIdx.y;   // 0..7  (FRAME_T slot)
  const int bh   = blockIdx.z;   // 0..47
  const int bht  = bh * T_ + tg;
  const int p0   = pblk * ROWS;
  const int hs   = lane >> 4;    // lane half (WMMA layout)
  const int lm   = lane & 15;

  const size_t att_off = ((size_t)bh * P_ + p0) * P_ + (size_t)tg * D_;

#if USE_TDM
  const uint32_t lds_logits = (uint32_t)(uintptr_t)(&logits[0]);
  tdm_load_tile(att_s + att_off, lds_logits);
#endif

  // Accumulators: after both phases C = o_s - o_t  (normalization folded into A weights)
  v8f C[2][4];
#pragma unroll
  for (int mb = 0; mb < 2; ++mb)
#pragma unroll
    for (int nb = 0; nb < 4; ++nb) {
      v8f z = {0.f, 0.f, 0.f, 0.f, 0.f, 0.f, 0.f, 0.f};
      C[mb][nb] = z;
    }

#pragma unroll 1
  for (int ph = 0; ph < 2; ++ph) {
#if USE_TDM
    __builtin_amdgcn_s_wait_tensorcnt(0);
#else
    {  // fallback staging: each lane copies its own row into LDS
      const float* attp = ph ? att_t : att_s;
      const float* grow = attp + att_off + (size_t)lane * P_;
#pragma unroll 2
      for (int j = 0; j < 49; ++j) {
        v4f q = *reinterpret_cast<const v4f*>(grow + 4 * j);
        *reinterpret_cast<v4f*>(&logits[lane * D_ + 4 * j]) = q;
      }
    }
#endif

    // ---- per-lane top-10 selection over this lane's 196 logits (branchless) ----
    float tk[10];
#pragma unroll
    for (int i = 0; i < 10; ++i) tk[i] = -3.0e38f;
#pragma unroll 2
    for (int j = 0; j < 49; ++j) {
      v4f q = *reinterpret_cast<const v4f*>(&logits[lane * D_ + 4 * j]);
#pragma unroll
      for (int c = 0; c < 4; ++c) {
        float x = q[c];
#pragma unroll
        for (int i = 0; i < 10; ++i) {  // max/min insertion network (VOPD-friendly)
          float hi = fmaxf(tk[i], x);
          x = fminf(tk[i], x);
          tk[i] = hi;
        }
      }
    }
    const float m0 = tk[0];  // row max
    const float mg = tk[9];  // 10th largest = mask margin (ties kept, matches reference)

    // ---- masked exp, write f16 weights into A buffer, accumulate norm ----
    float norm = 0.0f;
#pragma unroll 2
    for (int j = 0; j < 49; ++j) {
      v4f q = *reinterpret_cast<const v4f*>(&logits[lane * D_ + 4 * j]);
      v4h w;
#pragma unroll
      for (int c = 0; c < 4; ++c) {
        float wx = (q[c] >= mg) ? __expf(q[c] - m0) : 0.0f;
        norm += wx;
        w[c] = (h16)wx;
      }
      *reinterpret_cast<v4h*>(&abuf[lane * ASTR + 4 * j]) = w;
    }
    {  // zero K-padding [196, 224)
      v4h z = {(h16)0.f, (h16)0.f, (h16)0.f, (h16)0.f};
#pragma unroll
      for (int j = 49; j < 56; ++j)
        *reinterpret_cast<v4h*>(&abuf[lane * ASTR + 4 * j]) = z;
    }

#if USE_TDM
    if (ph == 0) tdm_load_tile(att_t + att_off, lds_logits);  // overlaps s-phase WMMA
#endif

    // ---- fold 1/norm (negated for phase t) into the A weights ----
    const float sgn = ph ? -1.0f : 1.0f;
    const h16 invh = (h16)(sgn / norm);
    const v4h invv = {invh, invh, invh, invh};
#pragma unroll 2
    for (int j = 0; j < 49; ++j) {
      v4h w = *reinterpret_cast<const v4h*>(&abuf[lane * ASTR + 4 * j]);
      w = w * invv;
      *reinterpret_cast<v4h*>(&abuf[lane * ASTR + 4 * j]) = w;
    }

    // ---- WMMA: C += A(16x32 f16, from LDS) x B(32x16 f16, L2-resident), K = 224 ----
    const h16* __restrict__ vb = (ph ? vh_t : vh_s) + (size_t)bht * E_ * D_;
#pragma unroll 1
    for (int ks = 0; ks < 7; ++ks) {
      v16h A[2];
#pragma unroll
      for (int mb = 0; mb < 2; ++mb) {
        int row = mb * 16 + lm;
        // A 16x32 layout (matches B): VGPR0-7 = K 16*hs..16*hs+15 contiguous
        v8h lo = *reinterpret_cast<const v8h*>(&abuf[row * ASTR + ks * 32 + 16 * hs]);
        v8h hi = *reinterpret_cast<const v8h*>(&abuf[row * ASTR + ks * 32 + 16 * hs + 8]);
        A[mb] = __builtin_shufflevector(lo, hi, 0, 1, 2, 3, 4, 5, 6, 7,
                                        8, 9, 10, 11, 12, 13, 14, 15);
      }
#pragma unroll
      for (int nb = 0; nb < 4; ++nb) {
        int n = nb * 16 + lm;
        v16h Bt;
#pragma unroll
        for (int r = 0; r < 8; ++r) {
          // 16-bit B 32x16 layout (contiguous K, matches A): VGPR r = K pair (2r + 16*hs), N = lane%16
          int d0 = ks * 32 + 2 * r + 16 * hs;
          if (d0 > D_ - 2) d0 = D_ - 2;  // clamp into real range; A is zero there
          v2h pr = *reinterpret_cast<const v2h*>(vb + (size_t)n * D_ + d0);
          Bt[2 * r] = pr[0];
          Bt[2 * r + 1] = pr[1];
        }
#pragma unroll
        for (int mb = 0; mb < 2; ++mb)
          C[mb][nb] = __builtin_amdgcn_wmma_f32_16x16x32_f16(
              false, A[mb], false, Bt, (short)0, C[mb][nb], false, false);
      }
    }
  }

  // ---- loss: C holds (o_s - o_t); sum of squares, wave-reduce, atomic add ----
  float acc = 0.0f;
#pragma unroll
  for (int mb = 0; mb < 2; ++mb)
#pragma unroll
    for (int nb = 0; nb < 4; ++nb)
#pragma unroll
      for (int v = 0; v < 8; ++v) acc = fmaf(C[mb][nb][v], C[mb][nb][v], acc);

#pragma unroll
  for (int off = 16; off > 0; off >>= 1) acc += __shfl_xor(acc, off, 32);
  if (lane == 0) atomicAdd(out, acc * INV_COUNT);
}

extern "C" void kernel_launch(void* const* d_in, const int* in_sizes, int n_in,
                              void* d_out, int out_size, void* d_ws, size_t ws_size,
                              hipStream_t stream) {
  (void)in_sizes; (void)n_in; (void)out_size; (void)ws_size;
  const float* att_s = (const float*)d_in[0];
  const float* att_t = (const float*)d_in[1];
  const float* v_s   = (const float*)d_in[2];
  const float* v_t   = (const float*)d_in[3];
  float* out = (float*)d_out;

  const size_t n_vh = (size_t)(B_ * H_) * T_ * E_ * D_;  // 4,816,896 halves per tensor
  h16* vh_s = (h16*)d_ws;
  h16* vh_t = vh_s + n_vh;

  hipMemsetAsync(d_out, 0, sizeof(float), stream);  // atomic accumulation target

  dim3 cgrid((unsigned)((n_vh + 255) / 256));
  vconvert<<<cgrid, 256, 0, stream>>>(v_s, vh_s);
  vconvert<<<cgrid, 256, 0, stream>>>(v_t, vh_t);

  dim3 grid(PBLK, T_, B_ * H_);
  vtop_fused<<<grid, dim3(32), 0, stream>>>(att_s, att_t, vh_s, vh_t, out);
}
